// GPT_12841952215777
// MI455X (gfx1250) — matrix operand order repaired and checked by reading, not compiled, not run on hardware
//
#include <hip/hip_runtime.h>
#include <cstdint>
#include <cstddef>

// ---------------------------------------------------------------------------
// GPT-2 small forward on gfx1250 (CDNA5 / MI455X).
// All GEMMs via v_wmma_f32_16x16x32_bf16 (wave32). A-tiles are staged by the
// Tensor Data Mover (tensor_load_to_lds, TENSORcnt) with descriptor padding
// matching the LDS pitch; stages are double-buffered so TDM + weight loads
// overlap the WMMA burst. Weights prefetched with global_prefetch_b8.
// ~660 GFLOP total, matrix-pipe bound: weights (495MB f32) stream once
// (~21us @ 23.3TB/s), per-layer weight set (28MB) is L2-resident (192MB L2).
// ---------------------------------------------------------------------------

typedef __bf16 bf16;
typedef __attribute__((ext_vector_type(16))) __bf16 v16bf;
typedef __attribute__((ext_vector_type(8)))  __bf16 bf16x8;
typedef __attribute__((ext_vector_type(8)))  float  v8f;
typedef __attribute__((ext_vector_type(4)))  unsigned int v4u;
typedef __attribute__((ext_vector_type(8)))  int    v8i;
typedef __attribute__((ext_vector_type(4)))  int    v4i;

#define BB 2
#define TT 1024
#define CC 768
#define HH 12
#define DD 64
#define LL 12
#define VV 50257
#define MM (BB*TT)      /* 2048 */
#define FCC (4*CC)      /* 3072 */

__device__ __forceinline__ v16bf frag_cat(bf16x8 lo, bf16x8 hi) {
    v16bf r;
#pragma unroll
    for (int e = 0; e < 8; ++e) { r[e] = lo[e]; r[e + 8] = hi[e]; }
    return r;
}

__device__ __forceinline__ v8f wmma_bf16(v16bf a, v16bf b, v8f c) {
    // D = A(16x32 bf16) x B(32x16 bf16) + C(16x16 f32)
    return __builtin_amdgcn_wmma_f32_16x16x32_bf16(
        /*neg_a=*/false, a, /*neg_b=*/false, b,
        /*c_mod=*/(short)0, c, /*reuse_a=*/false, /*reuse_b=*/false);
}

// Raw LDS byte offset: for LDS flat addresses, addr[31:0] is the LDS offset
// (CDNA5 ISA 10.2 aperture mapping), so truncating the generic pointer works.
__device__ __forceinline__ uint32_t lds_off(const void* p) {
    return (uint32_t)(uintptr_t)p;
}

// ---------------------------------------------------------------------------
// TDM: load a 2D tile (tile_w x tile_h elements of 2 bytes) from a row-major
// tensor (row stride stride_e elements) into LDS at lds_addr, inserting
// 16B of LDS padding after every 128B row (-> 72-bf16 LDS pitch).
// D# layout per CDNA5 ISA 8.3/8.4 (group0 128b, group1 256b).
// ---------------------------------------------------------------------------
__device__ __forceinline__ void tdm_load_tile_bf16(
    uint32_t lds_addr, const void* gaddr,
    uint32_t tensor_w_e, uint32_t tensor_h,
    uint32_t stride_e, uint32_t tile_w_e, uint32_t tile_h)
{
    const uint64_t ga = (uint64_t)(uintptr_t)gaddr;
    v4u g0;
    g0[0] = 1u;                                   // count=1, user descriptor
    g0[1] = lds_addr;                             // lds_addr[31:0] (bytes)
    g0[2] = (uint32_t)(ga & 0xFFFFFFFFu);         // global_addr[31:0]
    g0[3] = (uint32_t)((ga >> 32) & 0x1FFFFFFu)   // global_addr[56:32]
          | (2u << 30);                           // type=2 ("image")
    v8i g1;
    g1[0] = (int)((1u << 16)                      // data_size=1 -> 2 bytes
                | (1u << 20)                      // pad_enable
                | (4u << 22)                      // pad_interval: 32 DWORDs (128B row)
                | (3u << 25));                    // pad_amount: 4 DWORDs (16B)
    g1[1] = (int)((tensor_w_e & 0xFFFFu) << 16);                    // tensor_dim0[15:0]
    g1[2] = (int)((tensor_w_e >> 16) | ((tensor_h & 0xFFFFu) << 16)); // dim0 hi | dim1 lo
    g1[3] = (int)((tensor_h >> 16) | (tile_w_e << 16));             // dim1 hi | tile_dim0
    g1[4] = (int)tile_h;                                            // tile_dim1 (tile_dim2=0)
    g1[5] = (int)stride_e;                                          // tensor_dim0_stride lo32
    g1[6] = 0;                                                      // stride0 hi | stride1 lo
    g1[7] = 0;
    v4i z4; z4[0] = 0; z4[1] = 0; z4[2] = 0; z4[3] = 0;
#if __clang_major__ >= 23
    v8i z8; 
#pragma unroll
    for (int e = 0; e < 8; ++e) z8[e] = 0;
    __builtin_amdgcn_tensor_load_to_lds(g0, g1, z4, z4, z8, 0);
#else
    __builtin_amdgcn_tensor_load_to_lds(g0, g1, z4, z4, 0);
#endif
}

// ---------------------------------------------------------------------------
// Embedding: x[m,c] = tok[ids[m],c] + pos[m%T,c] + tok[spk[m],c]  (f32)
// ---------------------------------------------------------------------------
__global__ __launch_bounds__(256) void embed_kernel(
    const int* __restrict__ ids, const int* __restrict__ spk,
    const float* __restrict__ tok, const float* __restrict__ pos,
    float* __restrict__ x)
{
    const int m = blockIdx.x;
    const int t = m % TT;
    const int id = ids[m];
    const int sp = spk[m];
    const float* te = tok + (size_t)id * CC;
    const float* se = tok + (size_t)sp * CC;
    const float* pe = pos + (size_t)t * CC;
    float* xr = x + (size_t)m * CC;
#pragma unroll
    for (int i = 0; i < 3; ++i) {
        const int c = threadIdx.x + i * 256;
        xr[c] = te[c] + pe[c] + se[c];
    }
}

// ---------------------------------------------------------------------------
// LayerNorm over C=768 -> bf16 (WMMA A-operand precision).
// ---------------------------------------------------------------------------
__global__ __launch_bounds__(256) void ln_kernel(
    const float* __restrict__ x, const float* __restrict__ g,
    const float* __restrict__ bta, bf16* __restrict__ out)
{
    const int m = blockIdx.x;
    const float* xr = x + (size_t)m * CC;
    float vals[3];
    float s = 0.f, s2 = 0.f;
#pragma unroll
    for (int i = 0; i < 3; ++i) {
        const float vv = xr[threadIdx.x + i * 256];
        vals[i] = vv; s += vv; s2 += vv * vv;
    }
#pragma unroll
    for (int off = 16; off > 0; off >>= 1) {
        s  += __shfl_xor(s,  off);
        s2 += __shfl_xor(s2, off);
    }
    __shared__ float rs[8], rs2[8];
    if ((threadIdx.x & 31) == 0) { rs[threadIdx.x >> 5] = s; rs2[threadIdx.x >> 5] = s2; }
    __syncthreads();
    float ts = 0.f, ts2 = 0.f;
#pragma unroll
    for (int i = 0; i < 8; ++i) { ts += rs[i]; ts2 += rs2[i]; }
    const float mean = ts * (1.0f / CC);
    const float var  = ts2 * (1.0f / CC) - mean * mean;
    const float rstd = rsqrtf(var + 1e-5f);
    bf16* orow = out + (size_t)m * CC;
#pragma unroll
    for (int i = 0; i < 3; ++i) {
        const int c = threadIdx.x + i * 256;
        orow[c] = (bf16)((vals[i] - mean) * rstd * g[c] + bta[c]);
    }
}

// ---------------------------------------------------------------------------
// WMMA GEMM:  Y[M,N] = act( A[M,K](bf16) @ W[N,K](f32->bf16)^T + bias )
// Workgroup tile 128x128, 8 waves in 4(M)x2(N), per-wave 32x64 (8 accums).
// BK=64, double-buffered: TDM stages next A tile while WMMAs run; B tile
// converted f32->bf16 through VGPRs one stage ahead. 16 WMMAs per barrier.
// LDS pitch 72 bf16 (144B, 16B-aligned, matches TDM pad descriptor).
// A/B fragments use contiguous k (16*laneHalf + s); D layout per CDNA5 ISA 7.12.2.
// ---------------------------------------------------------------------------
template<bool GELU, bool RESID, bool OUTB, bool OUTF>
__global__ __launch_bounds__(256) void gemm_kernel(
    const bf16*  __restrict__ A,     // [M,K] bf16 row-major
    const float* __restrict__ W,     // [N,K] f32 row-major (torch Linear weight)
    const float* __restrict__ bias,  // [N] or nullptr
    bf16*  __restrict__ outB,        // [M,N] bf16 (if OUTB)
    float* __restrict__ outF,        // [M,N] f32  (if OUTF)
    int Ndim, int Kdim)
{
    __shared__ __align__(16) bf16 As[2][128][72];
    __shared__ __align__(16) bf16 Bs[2][128][72];

    const int tid   = threadIdx.x;
    const int lane  = tid & 31;
    const int wave  = tid >> 5;
    const int waveM = wave & 3;   // 0..3
    const int waveN = wave >> 2;  // 0..1
    const int mBlock = blockIdx.y * 128;
    const int nBlock = blockIdx.x * 128;

    v8f acc[2][4];
#pragma unroll
    for (int i = 0; i < 2; ++i)
#pragma unroll
        for (int j = 0; j < 4; ++j)
#pragma unroll
            for (int r = 0; r < 8; ++r) acc[i][j][r] = 0.0f;

    const int lr = tid >> 1;   // tile row 0..127
    const int lh = tid & 1;    // K half (32 elements each)

    int wr = nBlock + lr;
    if (wr >= Ndim) wr = Ndim - 1;            // tail clamp (head GEMM)
    const bf16*  atile_g = A + (size_t)mBlock * Kdim;   // block's A tile origin
    const float* wrow_g  = W + (size_t)wr * Kdim;

    // B tile stager: global f32 -> bf16 -> LDS (one stage ahead of compute)
    auto stageB = [&](int buf, int k0) {
        const float* src = wrow_g + k0 + lh * 32;
#pragma unroll
        for (int c = 0; c < 4; ++c) {
            const float4 f0 = *(const float4*)(src + c * 8);
            const float4 f1 = *(const float4*)(src + c * 8 + 4);
            bf16x8 bb;
            bb[0] = (bf16)f0.x; bb[1] = (bf16)f0.y; bb[2] = (bf16)f0.z; bb[3] = (bf16)f0.w;
            bb[4] = (bf16)f1.x; bb[5] = (bf16)f1.y; bb[6] = (bf16)f1.z; bb[7] = (bf16)f1.w;
            *(bf16x8*)&Bs[buf][lr][lh * 32 + c * 8] = bb;
        }
    };

    // --- prologue: stage k0=0 into buffer 0 ---
    if (wave == 0)
        tdm_load_tile_bf16(lds_off(&As[0][0][0]), atile_g,
                           (uint32_t)Kdim, (uint32_t)MM, (uint32_t)Kdim, 64u, 128u);
    stageB(0, 0);
    if (wave == 0) __builtin_amdgcn_s_wait_tensorcnt(0);
    __syncthreads();

    int buf = 0;
    for (int k0 = 0; k0 < Kdim; k0 += 64, buf ^= 1) {
        const int kn = k0 + 64;
        if (kn < Kdim) {
            // stage next K-tile into the alternate buffers (overlaps WMMAs)
            if (wave == 0)
                tdm_load_tile_bf16(lds_off(&As[buf ^ 1][0][0]), atile_g + kn,
                                   (uint32_t)Kdim, (uint32_t)MM, (uint32_t)Kdim, 64u, 128u);
            stageB(buf ^ 1, kn);
            if (kn + 64 < Kdim)
                __builtin_prefetch(wrow_g + kn + 64 + lh * 32, 0, 0);  // global_prefetch_b8
        }

        // --- two WMMA K-steps on current buffer ---
#pragma unroll
        for (int kk = 0; kk < 64; kk += 32) {
            const int arow = waveM * 32 + (lane & 15);
            const int koff = kk + (lane >> 4) * 16;  // A: lanes 0-15 K0-15, 16-31 K16-31
            v16bf af[2];
#pragma unroll
            for (int i = 0; i < 2; ++i) {
                bf16x8 lo = *(const bf16x8*)&As[buf][arow + i * 16][koff];
                bf16x8 hi = *(const bf16x8*)&As[buf][arow + i * 16][koff + 8];
                af[i] = frag_cat(lo, hi);
            }
            const int bko = kk + (lane >> 4) * 16;   // B: lanes 0-15 K0-15, 16-31 K16-31
            v16bf bfr[4];
#pragma unroll
            for (int j = 0; j < 4; ++j) {
                const int brow = waveN * 64 + j * 16 + (lane & 15);
                bf16x8 lo = *(const bf16x8*)&Bs[buf][brow][bko];
                bf16x8 hi = *(const bf16x8*)&Bs[buf][brow][bko + 8];
                bfr[j] = frag_cat(lo, hi);
            }
#pragma unroll
            for (int i = 0; i < 2; ++i)
#pragma unroll
                for (int j = 0; j < 4; ++j)
                    acc[i][j] = wmma_bf16(af[i], bfr[j], acc[i][j]);
        }

        // next stage must be fully resident before anyone reads it
        if (wave == 0) __builtin_amdgcn_s_wait_tensorcnt(0);
        __syncthreads();
    }

    // --- epilogue: D layout = row (VGPR r + 8*laneHalf), col (lane&15) ---
#pragma unroll
    for (int i = 0; i < 2; ++i) {
        const int row0 = mBlock + waveM * 32 + i * 16 + ((lane < 16) ? 0 : 8);
#pragma unroll
        for (int j = 0; j < 4; ++j) {
            const int col = nBlock + waveN * 64 + j * 16 + (lane & 15);
            if (col < Ndim) {
                const float bv = bias ? bias[col] : 0.0f;
#pragma unroll
                for (int r = 0; r < 8; ++r) {
                    float vl = acc[i][j][r] + bv;
                    if constexpr (GELU)
                        vl = 0.5f * vl * (1.0f + erff(vl * 0.70710678118f));
                    const size_t idx = (size_t)(row0 + r) * Ndim + col;
                    if constexpr (OUTF) {
                        if constexpr (RESID) outF[idx] += vl; else outF[idx] = vl;
                    }
                    if constexpr (OUTB) outB[idx] = (bf16)vl;
                }
            }
        }
    }
}

// ---------------------------------------------------------------------------
// Flash attention, one wave per (b, h, 16-row q tile); 32 keys per step.
// S = Q K^T via WMMA (D=64), online softmax with 16-lane shfl_xor
// reductions, P routed through LDS into A-layout, V staged LDS-transposed
// so P@V is 4 WMMAs per step. q,k,v,y layout: [B,T,H*D] bf16.
// ---------------------------------------------------------------------------
__global__ __launch_bounds__(128) void attn_kernel(
    const bf16* __restrict__ q, const bf16* __restrict__ k,
    const bf16* __restrict__ v, bf16* __restrict__ y)
{
    __shared__ __align__(16) bf16 Ps[4][16][40];
    __shared__ __align__(16) bf16 Vt[4][64][40];

    const int lane = threadIdx.x & 31;
    const int w    = threadIdx.x >> 5;
    const int wid  = blockIdx.x * 4 + w;
    const int qt   = wid & 63;        // T/16 = 64 q tiles
    const int bh   = wid >> 6;        // 0..23
    const int b    = bh / HH;
    const int h    = bh % HH;
    const int q0   = qt * 16;

    const size_t base = (size_t)b * TT * CC + (size_t)h * DD;
    const bf16* qp = q + base;
    const bf16* kp = k + base;
    const bf16* vp = v + base;

    // Q fragments (A layout), K-dim = 64 -> two 32-chunks
    const int qrow = q0 + (lane & 15);
    const int koff = (lane >> 4) * 16;
    v16bf aq[2];
#pragma unroll
    for (int c = 0; c < 2; ++c) {
        bf16x8 lo = *(const bf16x8*)(qp + (size_t)qrow * CC + c * 32 + koff);
        bf16x8 hi = *(const bf16x8*)(qp + (size_t)qrow * CC + c * 32 + koff + 8);
        aq[c] = frag_cat(lo, hi);
    }

    float mrow[8], lrow[8];
    v8f o[4];
#pragma unroll
    for (int r = 0; r < 8; ++r) { mrow[r] = -1e30f; lrow[r] = 0.0f; }
#pragma unroll
    for (int t = 0; t < 4; ++t)
#pragma unroll
        for (int r = 0; r < 8; ++r) o[t][r] = 0.0f;

    const int rbase = q0 + ((lane < 16) ? 0 : 8);
    const int jmax = (q0 + 15) >> 5;
    for (int j = 0; j <= jmax; ++j) {
        const int kb = j * 32;

        // stage V tile transposed: Vt[d][kk]; lane owns key row kb+lane
        {
            const bf16* vr = vp + (size_t)(kb + lane) * CC;
            bf16 tmp[64];
#pragma unroll
            for (int d8 = 0; d8 < 8; ++d8)
                *(bf16x8*)&tmp[d8 * 8] = *(const bf16x8*)(vr + d8 * 8);
#pragma unroll
            for (int d = 0; d < 64; ++d)
                Vt[w][d][lane] = tmp[d];
        }

        // scores: two 16x16 subtiles covering 32 keys
        float sc[2][8];
#pragma unroll
        for (int s = 0; s < 2; ++s) {
            v8f sacc;
#pragma unroll
            for (int r = 0; r < 8; ++r) sacc[r] = 0.0f;
#pragma unroll
            for (int c = 0; c < 2; ++c) {
                const int key = kb + s * 16 + (lane & 15);
                const int dof = c * 32 + (lane >> 4) * 16;
                bf16x8 lo = *(const bf16x8*)(kp + (size_t)key * CC + dof);
                bf16x8 hi = *(const bf16x8*)(kp + (size_t)key * CC + dof + 8);
                sacc = wmma_bf16(aq[c], frag_cat(lo, hi), sacc);
            }
            const int colg = kb + s * 16 + (lane & 15);
#pragma unroll
            for (int r = 0; r < 8; ++r) {
                float vv = sacc[r] * 0.125f;            // 1/sqrt(64)
                if (colg > rbase + r) vv = -1e30f;      // causal mask
                sc[s][r] = vv;
            }
        }

        // online softmax (rows live across 16 lanes of each half-wave)
#pragma unroll
        for (int r = 0; r < 8; ++r) {
            float rm = fmaxf(sc[0][r], sc[1][r]);
            rm = fmaxf(rm, __shfl_xor(rm, 1));
            rm = fmaxf(rm, __shfl_xor(rm, 2));
            rm = fmaxf(rm, __shfl_xor(rm, 4));
            rm = fmaxf(rm, __shfl_xor(rm, 8));
            const float newm  = fmaxf(mrow[r], rm);
            const float alpha = expf(mrow[r] - newm);
            mrow[r] = newm;
            const float p0 = expf(sc[0][r] - newm);
            const float p1 = expf(sc[1][r] - newm);
            sc[0][r] = p0; sc[1][r] = p1;
            float ps = p0 + p1;
            ps += __shfl_xor(ps, 1);
            ps += __shfl_xor(ps, 2);
            ps += __shfl_xor(ps, 4);
            ps += __shfl_xor(ps, 8);
            lrow[r] = lrow[r] * alpha + ps;
#pragma unroll
            for (int t = 0; t < 4; ++t) o[t][r] *= alpha;
        }

        // P (D layout) -> LDS -> A layout
        {
            const int prow = (lane < 16) ? 0 : 8;
#pragma unroll
            for (int s = 0; s < 2; ++s)
#pragma unroll
                for (int r = 0; r < 8; ++r)
                    Ps[w][prow + r][s * 16 + (lane & 15)] = (bf16)sc[s][r];
        }
        asm volatile("s_wait_dscnt 0" ::: "memory");

        v16bf pa;
        {
            const int pr = lane & 15;
            bf16x8 lo = *(const bf16x8*)&Ps[w][pr][koff];
            bf16x8 hi = *(const bf16x8*)&Ps[w][pr][koff + 8];
            pa = frag_cat(lo, hi);
        }
#pragma unroll
        for (int t = 0; t < 4; ++t) {
            const int vr = t * 16 + (lane & 15);
            const int vk = (lane >> 4) * 16;
            bf16x8 lo = *(const bf16x8*)&Vt[w][vr][vk];
            bf16x8 hi = *(const bf16x8*)&Vt[w][vr][vk + 8];
            o[t] = wmma_bf16(pa, frag_cat(lo, hi), o[t]);
        }
    }

    // write y (bf16) -- feeds proj GEMM as A operand
    bf16* yp = y + base;
#pragma unroll
    for (int t = 0; t < 4; ++t) {
        const int d = t * 16 + (lane & 15);
#pragma unroll
        for (int r = 0; r < 8; ++r) {
            const float ov = o[t][r] / lrow[r];
            yp[(size_t)(rbase + r) * CC + d] = (bf16)ov;
        }
    }
}

// ---------------------------------------------------------------------------
// Host-side orchestration
// ---------------------------------------------------------------------------
extern "C" void kernel_launch(void* const* d_in, const int* in_sizes, int n_in,
                              void* d_out, int out_size, void* d_ws, size_t ws_size,
                              hipStream_t stream)
{
    (void)in_sizes; (void)n_in; (void)out_size; (void)ws_size;

    const int*   input_ids   = (const int*)  d_in[0];
    const int*   speaker_ids = (const int*)  d_in[1];
    const float* tok_emb     = (const float*)d_in[2];
    const float* pos_emb     = (const float*)d_in[3];
    const float* ln1_w       = (const float*)d_in[4];
    const float* ln1_b       = (const float*)d_in[5];
    const float* wq          = (const float*)d_in[6];
    const float* bq          = (const float*)d_in[7];
    const float* wk          = (const float*)d_in[8];
    const float* bk          = (const float*)d_in[9];
    const float* wv          = (const float*)d_in[10];
    const float* bv          = (const float*)d_in[11];
    const float* wp          = (const float*)d_in[12];
    const float* bp          = (const float*)d_in[13];
    const float* ln2_w       = (const float*)d_in[14];
    const float* ln2_b       = (const float*)d_in[15];
    const float* wfc         = (const float*)d_in[16];
    const float* bfc         = (const float*)d_in[17];
    const float* wo          = (const float*)d_in[18];
    const float* bo          = (const float*)d_in[19];
    const float* lnf_w       = (const float*)d_in[20];
    const float* lnf_b       = (const float*)d_in[21];
    const float* head_w      = (const float*)d_in[22];

    // workspace carve-up (~34.6 MB)
    char* ws = (char*)d_ws;
    float* x  = (float*)ws;  ws += (size_t)MM * CC * sizeof(float);
    bf16* hb  = (bf16*)ws;   ws += (size_t)MM * CC * sizeof(bf16);
    bf16* qb  = (bf16*)ws;   ws += (size_t)MM * CC * sizeof(bf16);
    bf16* kb  = (bf16*)ws;   ws += (size_t)MM * CC * sizeof(bf16);
    bf16* vb  = (bf16*)ws;   ws += (size_t)MM * CC * sizeof(bf16);
    bf16* yb  = (bf16*)ws;   ws += (size_t)MM * CC * sizeof(bf16);
    bf16* fcb = (bf16*)ws;   ws += (size_t)MM * FCC * sizeof(bf16);

    embed_kernel<<<MM, 256, 0, stream>>>(input_ids, speaker_ids, tok_emb, pos_emb, x);

    const dim3 gq(CC / 128, MM / 128);    // 6 x 16
    const dim3 gf(FCC / 128, MM / 128);   // 24 x 16
    const int attnBlocks = (MM / 16) * HH / 4;  // 1536 waves / 4 per block

    for (int l = 0; l < LL; ++l) {
        const size_t oCC = (size_t)l * CC * CC;
        const size_t oFC = (size_t)l * FCC * CC;

        ln_kernel<<<MM, 256, 0, stream>>>(x, ln1_w + l * CC, ln1_b + l * CC, hb);
        gemm_kernel<false,false,true,false><<<gq, 256, 0, stream>>>(
            hb, wq + oCC, bq + l * CC, qb, nullptr, CC, CC);
        gemm_kernel<false,false,true,false><<<gq, 256, 0, stream>>>(
            hb, wk + oCC, bk + l * CC, kb, nullptr, CC, CC);
        gemm_kernel<false,false,true,false><<<gq, 256, 0, stream>>>(
            hb, wv + oCC, bv + l * CC, vb, nullptr, CC, CC);
        attn_kernel<<<attnBlocks, 128, 0, stream>>>(qb, kb, vb, yb);
        gemm_kernel<false,true,false,true><<<gq, 256, 0, stream>>>(
            yb, wp + oCC, bp + l * CC, nullptr, x, CC, CC);
        ln_kernel<<<MM, 256, 0, stream>>>(x, ln2_w + l * CC, ln2_b + l * CC, hb);
        gemm_kernel<true,false,true,false><<<gf, 256, 0, stream>>>(
            hb, wfc + oFC, bfc + l * FCC, fcb, nullptr, FCC, CC);
        gemm_kernel<false,true,false,true><<<gq, 256, 0, stream>>>(
            fcb, wo + oFC, bo + l * CC, nullptr, x, CC, FCC);
    }

    ln_kernel<<<MM, 256, 0, stream>>>(x, lnf_w, lnf_b, hb);
    const dim3 gh((VV + 127) / 128, MM / 128);  // 393 x 16, tail-guarded
    gemm_kernel<false,false,false,true><<<gh, 256, 0, stream>>>(
        hb, head_w, nullptr, nullptr, (float*)d_out, VV, CC);
}
